// C_SCNN_17918603559428
// MI455X (gfx1250) — matrix operand order repaired and checked by reading, not compiled, not run on hardware
//
#include <hip/hip_runtime.h>

typedef __attribute__((ext_vector_type(16))) __bf16 v16bf;
typedef __attribute__((ext_vector_type(8)))  __bf16 v8bf;
typedef __attribute__((ext_vector_type(8)))  float  v8f;

#define B_   4
#define C_   256
#define H_   64
#define W_   128
#define NC   16      // columns (b,w) per workgroup = WMMA N
#define NTH  512     // 16 waves, each owns one 16-channel M-tile
#define KCH  8       // 256 / 32 K-chunks
#define NPAD 40      // padded halfs per column row in LDS (16B aligned, bank-spread)
#define BROW (16 * NPAD)

__device__ __forceinline__ __bf16 f2bf(float f) {
  unsigned u = __builtin_bit_cast(unsigned, f);
  u += 0x7FFFu + ((u >> 16) & 1u);            // round-to-nearest-even
  unsigned short s = (unsigned short)(u >> 16);
  return __builtin_bit_cast(__bf16, s);
}

// -------------------------------------------------------------------------
// Kernel 1: both SCNN recurrence passes. One WG = 16 columns, all channels,
// all H steps of both passes (columns are independent across the whole
// double-pass, so no inter-block synchronization is ever needed).
// A (Wc) fragments live in VGPRs for the whole kernel; B (prev row) cycles
// through double-buffered LDS stored directly in WMMA B-fragment layout.
// Two independent accumulator chains (even/odd K-chunks) halve the serial
// WMMA RAW depth per step -- the step latency is the global limiter.
// -------------------------------------------------------------------------
__global__ __launch_bounds__(NTH) void scnn_double_pass(
    const float* __restrict__ xin, const float* __restrict__ wud,
    float* __restrict__ tmp1, float* __restrict__ tmp2)
{
  __shared__ __align__(16) __bf16 sP[2][KCH * BROW];

  const int tid   = threadIdx.x;
  const int lane  = tid & 31;
  const int wv    = tid >> 5;          // 0..15
  const int n     = lane & 15;         // N index (column) / M row for A
  const int g     = lane >> 4;         // lane half-group
  const int obase = wv * 16;           // this wave's output-channel tile
  const int b     = blockIdx.x >> 3;   // 4 batches
  const int w0    = (blockIdx.x & 7) * NC;

  // ---- preload Wc as bf16 A-fragments (constant across all 126 steps) ----
  // A 16x32 in this kernel's k order (same as B in LDS): lane m = lane%16; element e:
  //   K = g*16 + e
  v16bf aF[KCH];
  for (int kc = 0; kc < KCH; ++kc) {
    v16bf a;
#pragma unroll
    for (int e = 0; e < 16; ++e) {
      const int v  = e >> 1, hb = e & 1;
      const int kl = (g << 4) + e; (void)v; (void)hb;
      const int c  = kc * 32 + kl;
      a[e] = f2bf(wud[(size_t)((obase + n) * C_ + c) * 9 + 4]); // center tap
    }
    aF[kc] = a;
  }

  // Where this wave's 8 result channels land in the B-fragment LDS layout
  const int kcW = wv >> 1;                       // channel block / 32
  const int klW = ((wv & 1) << 4) + (g << 3);    // K_local base (8 contiguous)

  for (int pass = 0; pass < 2; ++pass) {
    const float* src = pass ? tmp1 : xin;
    float*       dst = pass ? tmp2 : tmp1;

    // step 0: row 0 copies straight through; goes to output row H-1 (reversal)
    for (int idx = tid; idx < C_ * NC; idx += NTH) {
      const int o  = idx >> 4;
      const int nn = idx & 15;
      const float v0 = src[((size_t)(b * C_ + o) * H_ + 0) * W_ + w0 + nn];
      dst[((size_t)(b * C_ + o) * H_ + (H_ - 1)) * W_ + w0 + nn] = v0;
      sP[0][(o >> 5) * BROW + nn * NPAD + (o & 31)] = f2bf(v0);
    }
    __syncthreads();

    int buf = 0;
    for (int i = 1; i < H_; ++i) {
      if (i + 1 < H_) {   // pull the next source row toward the WGP early
        __builtin_prefetch(
            &src[((size_t)(b * C_ + obase + (lane >> 1)) * H_ + i + 1) * W_ + w0], 0, 1);
      }
      // D = Wc @ prev  (16x16 tile per wave, K=256 over 8 bf16 WMMAs,
      // split into two independent accumulation chains)
      v8f acc0 = {0.f, 0.f, 0.f, 0.f, 0.f, 0.f, 0.f, 0.f};
      v8f acc1 = {0.f, 0.f, 0.f, 0.f, 0.f, 0.f, 0.f, 0.f};
#pragma unroll
      for (int kp = 0; kp < KCH / 2; ++kp) {
        const __bf16* p0 = &sP[buf][(2 * kp)     * BROW + n * NPAD + (g << 4)];
        const __bf16* p1 = &sP[buf][(2 * kp + 1) * BROW + n * NPAD + (g << 4)];
        v8bf lo0 = *(const v8bf*)(p0);
        v8bf hi0 = *(const v8bf*)(p0 + 8);
        v8bf lo1 = *(const v8bf*)(p1);
        v8bf hi1 = *(const v8bf*)(p1 + 8);
        v16bf bF0 = __builtin_shufflevector(lo0, hi0,
                      0,1,2,3,4,5,6,7,8,9,10,11,12,13,14,15);
        v16bf bF1 = __builtin_shufflevector(lo1, hi1,
                      0,1,2,3,4,5,6,7,8,9,10,11,12,13,14,15);
        acc0 = __builtin_amdgcn_wmma_f32_16x16x32_bf16(
                   false, aF[2 * kp],     false, bF0, (short)0, acc0, false, false);
        acc1 = __builtin_amdgcn_wmma_f32_16x16x32_bf16(
                   false, aF[2 * kp + 1], false, bF1, (short)0, acc1, false, false);
      }
      // cur = x_i + relu(D); write f32 row (reversed) + bf16 next-step operand
      const int nb = buf ^ 1;
      v8bf pk;
#pragma unroll
      for (int r = 0; r < 8; ++r) {
        const int o = obase + (g << 3) + r;        // C/D layout: M = r + 8g
        const float xi = src[((size_t)(b * C_ + o) * H_ + i) * W_ + w0 + n];
        const float cur = xi + fmaxf(acc0[r] + acc1[r], 0.f);
        dst[((size_t)(b * C_ + o) * H_ + (H_ - 1 - i)) * W_ + w0 + n] = cur;
        pk[r] = f2bf(cur);
      }
      *(v8bf*)&sP[nb][kcW * BROW + n * NPAD + klW] = pk;
      buf = nb;
      __syncthreads();
    }
    __threadfence();      // pass-1 global writes must be visible to all waves
    __syncthreads();
  }
}

// -------------------------------------------------------------------------
// Kernel 2: channel projection z[b,h,w] = sum_c W1[c]*x[b,c,h,w].
// (Projection commutes with the linear interpolation -> do it pre-upsample.)
// -------------------------------------------------------------------------
__global__ __launch_bounds__(256) void proj_kernel(
    const float* __restrict__ xin, const float* __restrict__ w1,
    float* __restrict__ z)
{
  __shared__ float red[256];
  const int blk  = blockIdx.x;           // 4*64*4 = 1024 blocks
  const int wch  = blk & 3;
  const int h    = (blk >> 2) & 63;
  const int b    = blk >> 8;
  const int wlo  = wch * 32;
  const int lw   = threadIdx.x & 31;
  const int cg   = threadIdx.x >> 5;     // 8 channel groups of 32
  float s = 0.f;
  for (int c = cg * 32; c < cg * 32 + 32; ++c)
    s += w1[c] * xin[((size_t)(b * C_ + c) * H_ + h) * W_ + wlo + lw];
  red[threadIdx.x] = s;
  __syncthreads();
  if (cg == 0) {
    float t = 0.f;
    for (int k = 0; k < 8; ++k) t += red[k * 32 + lw];
    z[(size_t)(b * H_ + h) * W_ + wlo + lw] = t;
  }
}

// -------------------------------------------------------------------------
// Kernel 3: bilinear upsample (align_corners, H then W order as reference)
// + deterministic per-block partial sums for mean/var.
// -------------------------------------------------------------------------
__global__ __launch_bounds__(256) void interp_kernel(
    const float* __restrict__ z, float* __restrict__ yup,
    float* __restrict__ part)
{
  __shared__ float s1[256], s2[256];
  const int t   = threadIdx.x;
  const int idx = blockIdx.x * 256 + t;        // 0..524287
  const int ww  = idx & 511;
  const int hh  = (idx >> 9) & 255;
  const int b   = idx >> 17;
  const float ph = hh * (63.0f / 255.0f);
  const float pw = ww * (127.0f / 511.0f);
  int hlo = (int)floorf(ph); hlo = hlo < 0 ? 0 : (hlo > 63 ? 63 : hlo);
  int hhi = hlo + 1 > 63 ? 63 : hlo + 1;
  int wlo = (int)floorf(pw); wlo = wlo < 0 ? 0 : (wlo > 127 ? 127 : wlo);
  int whi = wlo + 1 > 127 ? 127 : wlo + 1;
  const float fh = ph - (float)hlo;
  const float fw = pw - (float)wlo;
  const float* zb = z + (size_t)b * H_ * W_;
  const float a00 = zb[hlo * W_ + wlo], a01 = zb[hlo * W_ + whi];
  const float a10 = zb[hhi * W_ + wlo], a11 = zb[hhi * W_ + whi];
  const float c0 = a00 + (a10 - a00) * fh;     // H interp first
  const float c1 = a01 + (a11 - a01) * fh;
  const float y  = c0 + (c1 - c0) * fw;        // then W
  yup[idx] = y;
  s1[t] = y; s2[t] = y * y;
  __syncthreads();
  for (int off = 128; off; off >>= 1) {
    if (t < off) { s1[t] += s1[t + off]; s2[t] += s2[t + off]; }
    __syncthreads();
  }
  if (t == 0) { part[2 * blockIdx.x] = s1[0]; part[2 * blockIdx.x + 1] = s2[0]; }
}

// -------------------------------------------------------------------------
// Kernel 4: single-block deterministic reduction -> mean, rsqrt(var+eps)
// -------------------------------------------------------------------------
__global__ __launch_bounds__(256) void stats_kernel(
    const float* __restrict__ part, float* __restrict__ stats)
{
  __shared__ float s1[256], s2[256];
  const int t = threadIdx.x;
  float a = 0.f, q = 0.f;
  for (int k = t; k < 2048; k += 256) { a += part[2 * k]; q += part[2 * k + 1]; }
  s1[t] = a; s2[t] = q;
  __syncthreads();
  for (int off = 128; off; off >>= 1) {
    if (t < off) { s1[t] += s1[t + off]; s2[t] += s2[t + off]; }
    __syncthreads();
  }
  if (t == 0) {
    const float N = 524288.0f;
    const float mean = s1[0] / N;
    float var = s2[0] / N - mean * mean;
    if (var < 0.f) var = 0.f;
    stats[0] = mean;
    stats[1] = rsqrtf(var + 1e-5f);
  }
}

// -------------------------------------------------------------------------
// Kernel 5: normalize + affine + sigmoid
// -------------------------------------------------------------------------
__global__ __launch_bounds__(256) void finalize_kernel(
    const float* __restrict__ yup, const float* __restrict__ stats,
    const float* __restrict__ gamma, const float* __restrict__ beta,
    float* __restrict__ out)
{
  const int idx = blockIdx.x * 256 + threadIdx.x;
  const float m = stats[0], rs = stats[1];
  const float yv = (yup[idx] - m) * rs * gamma[0] + beta[0];
  out[idx] = 1.0f / (1.0f + expf(-yv));
}

extern "C" void kernel_launch(void* const* d_in, const int* in_sizes, int n_in,
                              void* d_out, int out_size, void* d_ws, size_t ws_size,
                              hipStream_t stream) {
  const float* p2c   = (const float*)d_in[0];
  const float* wud   = (const float*)d_in[1];
  const float* w1    = (const float*)d_in[2];
  const float* gamma = (const float*)d_in[3];
  const float* beta  = (const float*)d_in[4];
  float* out = (float*)d_out;

  float* ws   = (float*)d_ws;
  float* tmp1 = ws;                 // [4,256,64,128] f32 (pass-1 history)
  float* tmp2 = ws + 8388608;       // [4,256,64,128] f32 (pass-2 output)
  // tmp1 region is dead after kernel 1 -> reuse for the small buffers:
  float* z     = ws;                         // [4,64,128]
  float* yup   = ws + 32768;                 // [4,256,512]
  float* part  = ws + 32768 + 524288;        // 2048 * {sum, sumsq}
  float* stats = ws + 32768 + 524288 + 4096; // {mean, rsqrt}

  hipLaunchKernelGGL(scnn_double_pass, dim3(32),   dim3(NTH), 0, stream, p2c, wud, tmp1, tmp2);
  hipLaunchKernelGGL(proj_kernel,      dim3(1024), dim3(256), 0, stream, tmp2, w1, z);
  hipLaunchKernelGGL(interp_kernel,    dim3(2048), dim3(256), 0, stream, z, yup, part);
  hipLaunchKernelGGL(stats_kernel,     dim3(1),    dim3(256), 0, stream, part, stats);
  hipLaunchKernelGGL(finalize_kernel,  dim3(2048), dim3(256), 0, stream, yup, stats, gamma, beta, out);
}
